// AttentionHead_61289183314310
// MI455X (gfx1250) — matrix operand order repaired and checked by reading, not compiled, not run on hardware
//
#include <hip/hip_runtime.h>

// ---- problem constants (match reference) ----
#define B_   4
#define L_   4096
#define D_   1024
#define HS   64      // head size
#define NCOL 192     // 3*HS

typedef __bf16 bf16_t;
typedef __attribute__((ext_vector_type(16))) __bf16 v16bf;
typedef __attribute__((ext_vector_type(8)))  float  v8f;

// CDNA5 async global->LDS copy (ASYNCcnt-tracked). If the assembler rejects
// the mnemonic, set to 0 to fall back to regular load+ds_store staging.
#define USE_ASYNC_LDS 1

__device__ __forceinline__ void async_ld16(unsigned lds_off, unsigned long long gaddr) {
#if USE_ASYNC_LDS
    asm volatile("global_load_async_to_lds_b128 %0, %1, off"
                 :: "v"(lds_off), "v"(gaddr) : "memory");
#else
    const float4 v = *(const float4*)(uintptr_t)gaddr;
    // reconstruct a generic shared pointer from the 32-bit LDS offset
    extern __shared__ char __lds_base[];
    *(float4*)(__lds_base + (lds_off - (unsigned)(uintptr_t)__lds_base)) = v;
#endif
}

__device__ __forceinline__ void wait_async0() {
#if USE_ASYNC_LDS
    asm volatile("s_wait_asynccnt 0x0" ::: "memory");
#endif
}

// =====================================================================
// Kernel 0: transpose + convert W (D x 192 f32) -> Wt (192 x D bf16)
// =====================================================================
__global__ __launch_bounds__(256)
void wt_kernel(const float* __restrict__ w, bf16_t* __restrict__ Wt) {
    int idx = blockIdx.x * 256 + threadIdx.x;      // 0 .. 192*1024-1
    int c = idx / D_;
    int k = idx % D_;
    Wt[(size_t)c * D_ + k] = (bf16_t)w[(size_t)k * NCOL + c];
}

// =====================================================================
// Kernel 1: QKV projection  (B*L, D) x (D, 192) with bf16 WMMA.
// Q,K row-major bf16; V stored transposed (B, 64, L) bf16.
// =====================================================================
__global__ __launch_bounds__(128)
void qkv_proj_kernel(const float* __restrict__ x, const bf16_t* __restrict__ Wt,
                     bf16_t* __restrict__ Qb, bf16_t* __restrict__ Kb,
                     bf16_t* __restrict__ Vt) {
    const int lane = threadIdx.x & 31;
    const int wid  = threadIdx.x >> 5;
    const int half = lane >> 4;
    const int l16  = lane & 15;
    const int tm   = blockIdx.x * 4 + wid;   // row tile (0..1023)
    const int tn   = blockIdx.y;             // col tile (0..11), uniform

    const int row = tm * 16 + l16;
    const int col = tn * 16 + l16;

    v8f acc = {};
    for (int k0 = 0; k0 < D_; k0 += 32) {
        const float* xa = x + (size_t)row * D_ + k0 + (half ? 16 : 0);
        v16bf a;
#pragma unroll
        for (int i = 0; i < 8; ++i) {
            a[i]     = (bf16_t)xa[i];
            a[i + 8] = (bf16_t)xa[8 + i];
        }
        const bf16_t* wb = Wt + (size_t)col * D_ + k0 + (half ? 16 : 0);
        v16bf b;
#pragma unroll
        for (int i = 0; i < 16; ++i) b[i] = wb[i];

        acc = __builtin_amdgcn_wmma_f32_16x16x32_bf16(
                  false, a, false, b, (short)0, acc, false, false);
    }

    const int row0 = tm * 16 + (half ? 8 : 0);
    if (tn < 4) {
#pragma unroll
        for (int j = 0; j < 8; ++j)
            Qb[(size_t)(row0 + j) * HS + col] = (bf16_t)acc[j];
    } else if (tn < 8) {
        const int kc = col - HS;
#pragma unroll
        for (int j = 0; j < 8; ++j)
            Kb[(size_t)(row0 + j) * HS + kc] = (bf16_t)acc[j];
    } else {
        const int vc  = col - 2 * HS;
        const int bi  = (tm * 16) / L_;
        const int li0 = row0 - bi * L_;
        bf16_t* vdst = Vt + ((size_t)bi * HS + vc) * L_ + li0;
#pragma unroll
        for (int j = 0; j < 8; ++j) vdst[j] = (bf16_t)acc[j];
    }
}

// =====================================================================
// Kernel 2: flash attention, 4 waves/block (same batch, 4 query tiles).
// Double-buffered async K/V staging in LDS shared by all waves:
//   per 64-key tile: 8 KB K-block + 8 KB V^T-block via
//   global_load_async_to_lds_b128, s_wait_asynccnt handshake.
// Per tile: 8 WMMA for S, online softmax (16-lane butterflies),
// P via LDS into A-fragment layout, 8 chained WMMA for O += P*V.
// =====================================================================
__global__ __launch_bounds__(128)
void attn_kernel(const bf16_t* __restrict__ Qb, const bf16_t* __restrict__ Kb,
                 const bf16_t* __restrict__ Vt, const unsigned char* __restrict__ mask,
                 float* __restrict__ out) {
    __shared__ bf16_t kvbuf[2][2][64 * 64];  // [buf][K/V][8 KB]  = 32 KB
    __shared__ bf16_t pshm[4][16 * 64];      // per-wave P staging =  8 KB

    const int tid  = threadIdx.x;
    const int lane = tid & 31;
    const int wid  = tid >> 5;
    const int half = lane >> 4;
    const int l16  = lane & 15;

    // blockIdx.x: 0..255; all 4 waves of a block share one batch
    const int tile = blockIdx.x * 4 + wid;    // 0..1023
    const int b    = tile / (L_ / 16);
    const int q0   = (tile % (L_ / 16)) * 16;

    const bf16_t* Kbase = Kb + (size_t)b * L_ * HS;
    const bf16_t* Vbase = Vt + (size_t)b * HS * L_;
    const unsigned char* mbase = mask + (size_t)b * L_;

    // ---- Q fragments (global, once): two 16x32 bf16 A-operands
    const bf16_t* qrow = Qb + ((size_t)(b * L_ + q0 + l16)) * HS;
    v16bf qa[2];
#pragma unroll
    for (int f = 0; f < 2; ++f) {
        const bf16_t* p = qrow + f * 32 + (half ? 16 : 0);
#pragma unroll
        for (int i = 0; i < 8; ++i) { qa[f][i] = p[i]; qa[f][i + 8] = p[8 + i]; }
    }

    // ---- async staging of one 64-key tile into kvbuf[buf] ----
    // K block: 64 rows x 128 B contiguous.  V block: 64 dim-rows x 128 B.
    auto stage = [&](int buf, int j0) {
        const unsigned kl = (unsigned)(uintptr_t)&kvbuf[buf][0][0];
        const unsigned vl = (unsigned)(uintptr_t)&kvbuf[buf][1][0];
        const unsigned long long kg = (unsigned long long)(uintptr_t)(Kbase + (size_t)j0 * HS);
        const unsigned long long vg = (unsigned long long)(uintptr_t)(Vbase + j0);
#pragma unroll
        for (int i = 0; i < 4; ++i) {             // 512 x 16B K chunks
            const int c = i * 128 + tid;
            async_ld16(kl + c * 16, kg + (unsigned long long)c * 16);
        }
#pragma unroll
        for (int i = 0; i < 4; ++i) {             // 512 x 16B V chunks
            const int c = i * 128 + tid;
            const int row = c >> 3, off = (c & 7) * 16;
            async_ld16(vl + c * 16,
                       vg + (unsigned long long)row * (L_ * 2) + off);
        }
    };

    // ---- softmax running state; element j <-> query row j + half*8
    float m[8], l[8];
    v8f   o[4];
#pragma unroll
    for (int j = 0; j < 8; ++j) { m[j] = -1e30f; l[j] = 0.0f; }
#pragma unroll
    for (int dt = 0; dt < 4; ++dt) o[dt] = (v8f){};

    const float scale = 0.125f;   // 64^-0.5
    const int NIT = L_ / 64;

    stage(0, 0);

    for (int it = 0; it < NIT; ++it) {
        const int j0  = it * 64;
        const int buf = it & 1;

        wait_async0();
        __syncthreads();                       // staged data visible to all
        if (it + 1 < NIT) stage(buf ^ 1, j0 + 64);

        const bf16_t* Kl = &kvbuf[buf][0][0];
        const bf16_t* Vl = &kvbuf[buf][1][0];

        // ---- S = Q*K^T : four 16x16 f32 tiles
        v8f s[4];
#pragma unroll
        for (int t = 0; t < 4; ++t) {
            v8f acc = {};
#pragma unroll
            for (int c = 0; c < 2; ++c) {
                const bf16_t* kp = Kl + (t * 16 + l16) * HS + c * 32 + (half ? 16 : 0);
                v16bf kb;
#pragma unroll
                for (int i = 0; i < 16; ++i) kb[i] = kp[i];
                acc = __builtin_amdgcn_wmma_f32_16x16x32_bf16(
                          false, qa[c], false, kb, (short)0, acc, false, false);
            }
            s[t] = acc;
        }

        bool mk[4];
#pragma unroll
        for (int t = 0; t < 4; ++t) mk[t] = mbase[j0 + t * 16 + l16] != 0;

        // ---- online softmax (butterfly over the 16-lane group)
        float alpha[8];
#pragma unroll
        for (int j = 0; j < 8; ++j) {
            float bm = -1e30f;
#pragma unroll
            for (int t = 0; t < 4; ++t) {
                const float sv = mk[t] ? s[t][j] * scale : -1e30f;
                s[t][j] = sv;
                bm = fmaxf(bm, sv);
            }
#pragma unroll
            for (int off = 1; off < 16; off <<= 1)
                bm = fmaxf(bm, __shfl_xor(bm, off, 32));
            const float mn = fmaxf(m[j], bm);
            alpha[j] = __expf(m[j] - mn);
            m[j] = mn;
        }

        // ---- P = exp(S - m): stage to per-wave LDS + rowsum
        bf16_t* pl = pshm[wid];
#pragma unroll
        for (int j = 0; j < 8; ++j) {
            const int r = j + (half ? 8 : 0);
            float rs = 0.0f;
#pragma unroll
            for (int t = 0; t < 4; ++t) {
                const float p = mk[t] ? __expf(s[t][j] - m[j]) : 0.0f;
                pl[r * 64 + t * 16 + l16] = (bf16_t)p;
                rs += p;
            }
#pragma unroll
            for (int off = 1; off < 16; off <<= 1)
                rs += __shfl_xor(rs, off, 32);
            l[j] = l[j] * alpha[j] + rs;
        }
        __builtin_amdgcn_wave_barrier();       // DS in-order per wave

        v16bf pa[2];
#pragma unroll
        for (int f = 0; f < 2; ++f) {
            const bf16_t* pr = pl + l16 * 64 + f * 32 + (half ? 16 : 0);
#pragma unroll
            for (int i = 0; i < 8; ++i) { pa[f][i] = pr[i]; pa[f][i + 8] = pr[8 + i]; }
        }
        __builtin_amdgcn_wave_barrier();

        // ---- O = O*alpha + P*V
#pragma unroll
        for (int dt = 0; dt < 4; ++dt) {
            v8f c = o[dt];
#pragma unroll
            for (int j = 0; j < 8; ++j) c[j] *= alpha[j];
#pragma unroll
            for (int f = 0; f < 2; ++f) {
                const bf16_t* vp = Vl + (dt * 16 + l16) * 64 + f * 32 + (half ? 16 : 0);
                v16bf vb;
#pragma unroll
                for (int i = 0; i < 16; ++i) vb[i] = vp[i];
                c = __builtin_amdgcn_wmma_f32_16x16x32_bf16(
                        false, pa[f], false, vb, (short)0, c, false, false);
            }
            o[dt] = c;
        }

        __syncthreads();   // all waves done with kvbuf[buf] before reuse
    }

    // ---- finalize: out[b, q0+r, d] = O / l  (f32 output)
#pragma unroll
    for (int j = 0; j < 8; ++j) {
        const float inv = 1.0f / l[j];
        const int r = j + (half ? 8 : 0);
#pragma unroll
        for (int dt = 0; dt < 4; ++dt) {
            const int d = dt * 16 + l16;
            out[(size_t)(b * L_ + q0 + r) * HS + d] = o[dt][j] * inv;
        }
    }
}

// =====================================================================
// Host-side launch
// =====================================================================
extern "C" void kernel_launch(void* const* d_in, const int* in_sizes, int n_in,
                              void* d_out, int out_size, void* d_ws, size_t ws_size,
                              hipStream_t stream) {
    (void)in_sizes; (void)n_in; (void)out_size; (void)ws_size;

    const float*         x    = (const float*)d_in[0];
    const unsigned char* mask = (const unsigned char*)d_in[1]; // jnp.bool_ -> 1 byte/elem
    const float*         w    = (const float*)d_in[2];
    float*               out  = (float*)d_out;

    bf16_t* Qb = (bf16_t*)d_ws;
    bf16_t* Kb = Qb + (size_t)B_ * L_ * HS;
    bf16_t* Vt = Kb + (size_t)B_ * L_ * HS;
    bf16_t* Wt = Vt + (size_t)B_ * L_ * HS;

    wt_kernel<<<(NCOL * D_) / 256, 256, 0, stream>>>(w, Wt);

    dim3 g1(B_ * L_ / 64, NCOL / 16);
    qkv_proj_kernel<<<g1, 128, 0, stream>>>(x, Wt, Qb, Kb, Vt);

    attn_kernel<<<(B_ * L_ / 16) / 4, 128, 0, stream>>>(Qb, Kb, Vt, mask, out);
}
